// Autoformer_84619445666464
// MI455X (gfx1250) — matrix operand order repaired and checked by reading, not compiled, not run on hardware
//
#include <hip/hip_runtime.h>
#include <hip/hip_bf16.h>
#include <math.h>

// Problem constants (match reference)
#define BB    8
#define LL    3072
#define CC    1024      // D_MODEL = H*E
#define TOPK  8
#define MROWS (BB * LL) // 24576

typedef __attribute__((ext_vector_type(16))) __bf16 v16bf;
typedef __attribute__((ext_vector_type(8)))  float  v8f;

union FragBF { v16bf v; uint4 u[2]; };

// CDNA5 async copy to LDS (gfx1250): guarded so either toolchain compiles.
// Builtin signature (from clang diagnostics):
//   void __builtin_amdgcn_global_load_async_to_lds_b128(
//       v4i addrspace(1)* gsrc, v4i addrspace(3)* ldst, imm offset, imm cpol)
typedef int v4i __attribute__((__vector_size__(16)));
typedef __attribute__((address_space(1))) v4i* as1_v4i;
typedef __attribute__((address_space(3))) v4i* as3_v4i;

#if defined(__HIP_DEVICE_COMPILE__) && \
    __has_builtin(__builtin_amdgcn_global_load_async_to_lds_b128) && \
    __has_builtin(__builtin_amdgcn_s_wait_asynccnt)
#define HAVE_ASYNC 1
#define ASYNC_COPY16(dst, src)                                              \
  __builtin_amdgcn_global_load_async_to_lds_b128(                           \
      (as1_v4i)(void*)(src), (as3_v4i)(void*)(dst), 0, 0)
#define ASYNC_WAIT(n) __builtin_amdgcn_s_wait_asynccnt(n)
#else
#define HAVE_ASYNC 0
#define ASYNC_COPY16(dst, src) (*(uint4*)(dst) = *(const uint4*)(src))
#define ASYNC_WAIT(n)
#endif

// ---------------------------------------------------------------------------
// LDS fragment loaders, matching CDNA5 WMMA 16-bit VGPR layouts (ISA 7.12.2).
// A (16x32 bf16): lane<16 -> M=lane, K = {0..7, 16..23}; lane>=16 -> M=lane-16,
//                 K = {8..15, 24..31}. LDS A tile is [m][k], k contiguous.
// B (32x16 bf16): lane<16 -> N=lane, K = {0..7, 16..23}; lane>=16 -> N=lane-16, K = {8..15, 24..31}.
//                 LDS B tile is stored N-major [n][k] so 16 K's are contiguous.
// ---------------------------------------------------------------------------
__device__ __forceinline__ v16bf ld_frag_a(const __bf16* base, int lane, int ld) {
  const int row = lane & 15;
  const int kh  = (lane & 16) ? 8 : 0;
  const __bf16* p = base + row * ld + kh;
  FragBF f;
  f.u[0] = *(const uint4*)(p);        // K = kh .. kh+7
  f.u[1] = *(const uint4*)(p + 16);   // K = kh+16 .. kh+23
  return f.v;
}

__device__ __forceinline__ v16bf ld_frag_b(const __bf16* base, int lane, int ld) {
  const int col = lane & 15;
  const int kh  = (lane & 16) ? 8 : 0;
  const __bf16* p = base + col * ld + kh;
  FragBF f;
  f.u[0] = *(const uint4*)(p);        // K = kh .. kh+7
  f.u[1] = *(const uint4*)(p + 16);   // K = kh+16 .. kh+23
  return f.v;
}

// ---------------------------------------------------------------------------
// GEMM: C[M,N] = A[M,K] * B[K,N] + bias, fp32 in, fp32 or bf16 out.
// Block tile 128x128, 8 waves (wave32), wave tile 32x64 (2x4 WMMA tiles).
// Software pipeline: next tile's global loads prefetched into registers while
// WMMAs consume the current LDS buffer; double-buffered LDS, 1 barrier/step.
// fp32 -> bf16 conversion happens on the LDS fill path.
// ---------------------------------------------------------------------------
template <bool OUT_BF16>
__global__ __launch_bounds__(256) void gemm_bias(
    const float* __restrict__ A, const float* __restrict__ Bw,
    const float* __restrict__ bias, void* __restrict__ Cout,
    int M, int N, int K)
{
  __shared__ __bf16 As[2][128][40];   // [buf][m][k], +8 pad
  __shared__ __bf16 Bs[2][128][40];   // [buf][n][k] (transposed), +8 pad

  const int tid  = threadIdx.x;
  const int lane = tid & 31;
  const int wave = tid >> 5;
  const int wm   = wave & 3;       // 4 waves along M
  const int wn   = wave >> 2;      // 2 waves along N
  const int m0   = blockIdx.y * 128;
  const int n0   = blockIdx.x * 128;

  float4 ra[4], rb[4];

  auto g_load = [&](int kk) {
#pragma unroll
    for (int it = 0; it < 4; ++it) {
      const int linear = tid + it * 256;
      const int arow = linear >> 3, ac = (linear & 7) * 4;
      ra[it] = *(const float4*)&A[(size_t)(m0 + arow) * K + kk + ac];
      const int brow = linear >> 5, bc = (linear & 31) * 4;
      rb[it] = *(const float4*)&Bw[(size_t)(kk + brow) * N + n0 + bc];
    }
  };
  auto l_store = [&](int buf) {
#pragma unroll
    for (int it = 0; it < 4; ++it) {
      const int linear = tid + it * 256;
      const int arow = linear >> 3, ac = (linear & 7) * 4;
      As[buf][arow][ac + 0] = (__bf16)ra[it].x;
      As[buf][arow][ac + 1] = (__bf16)ra[it].y;
      As[buf][arow][ac + 2] = (__bf16)ra[it].z;
      As[buf][arow][ac + 3] = (__bf16)ra[it].w;
      const int brow = linear >> 5, bc = (linear & 31) * 4;
      Bs[buf][bc + 0][brow] = (__bf16)rb[it].x;
      Bs[buf][bc + 1][brow] = (__bf16)rb[it].y;
      Bs[buf][bc + 2][brow] = (__bf16)rb[it].z;
      Bs[buf][bc + 3][brow] = (__bf16)rb[it].w;
    }
  };

  const v8f vzero = {0.f, 0.f, 0.f, 0.f, 0.f, 0.f, 0.f, 0.f};
  v8f acc[2][4];
#pragma unroll
  for (int i = 0; i < 2; ++i)
#pragma unroll
    for (int j = 0; j < 4; ++j) acc[i][j] = vzero;

  // Pipeline prologue
  g_load(0);
  l_store(0);
  __syncthreads();

  int cur = 0;
  for (int kk = 0; kk < K; kk += 32) {
    const bool has_next = (kk + 32 < K);
    if (has_next) g_load(kk + 32);     // global loads in flight during WMMAs

    v16bf af[2], bfm[4];
    af[0] = ld_frag_a(&As[cur][wm * 32 +  0][0], lane, 40);
    af[1] = ld_frag_a(&As[cur][wm * 32 + 16][0], lane, 40);
#pragma unroll
    for (int j = 0; j < 4; ++j)
      bfm[j] = ld_frag_b(&Bs[cur][wn * 64 + j * 16][0], lane, 40);

#pragma unroll
    for (int i = 0; i < 2; ++i)
#pragma unroll
      for (int j = 0; j < 4; ++j)
        acc[i][j] = __builtin_amdgcn_wmma_f32_16x16x32_bf16(
            false, af[i], false, bfm[j], (short)0, acc[i][j], false, false);

    if (has_next) l_store(cur ^ 1);    // fill other buffer (no reader conflict)
    __syncthreads();
    cur ^= 1;
  }

  // Epilogue: C/D layout -> row = tile_m*16 + r + 8*(lane>=16), col = lane&15
#pragma unroll
  for (int i = 0; i < 2; ++i) {
    const int mb = m0 + wm * 32 + i * 16 + ((lane >> 4) << 3);
#pragma unroll
    for (int j = 0; j < 4; ++j) {
      const int ncol = n0 + wn * 64 + j * 16 + (lane & 15);
      const float bv = bias[ncol];
#pragma unroll
      for (int r = 0; r < 8; ++r) {
        const float  val = acc[i][j][r] + bv;
        const size_t off = (size_t)(mb + r) * N + ncol;
        if (OUT_BF16) ((__bf16*)Cout)[off] = (__bf16)val;
        else          ((float*) Cout)[off] = val;
      }
    }
  }
}

// ---------------------------------------------------------------------------
// Channel-averaged circular auto-correlation via WMMA Gram tiles:
//   corr[b,tau] += sum_{u,t: (u-t)%L==tau} <q[b,u,:], k[b,t,:]>
// One workgroup per (b, u-block of 64). Q/K are already bf16 -> tile fills are
// pure byte copies: use GLOBAL_LOAD_ASYNC_TO_LDS_B128 with double buffering
// (issue stage s+1, s_wait_asynccnt<=2 retires stage s, barrier, ds_load+wmma).
// Local tau-histogram in LDS (ds_add_f32), flushed once with global atomics.
// ---------------------------------------------------------------------------
__global__ __launch_bounds__(256) void corr_kernel(
    const __bf16* __restrict__ Qb, const __bf16* __restrict__ Kb,
    float* __restrict__ corr)
{
  __shared__ float  hist[LL];           // 12 KB
  __shared__ __bf16 qt[2][64][40];
  __shared__ __bf16 kt[2][64][40];

  const int b  = blockIdx.y;
  const int u0 = blockIdx.x * 64;
  const __bf16* Qbase = Qb + (size_t)b * LL * CC;
  const __bf16* Kbase = Kb + (size_t)b * LL * CC;

  const int tid  = threadIdx.x;
  const int lane = tid & 31;
  const int wave = tid >> 5;
  const int sm   = wave & 3;            // 4 waves along u (16 rows each)
  const int sn   = wave >> 2;           // 2 waves along t (32 cols each)

  for (int i = tid; i < LL; i += 256) hist[i] = 0.f;
  __syncthreads();

  const int lrow = tid >> 2;            // 64 rows, 4 x 16B per row
  const int lc   = (tid & 3) * 8;       // bf16 column offset

  auto stage_copy = [&](int buf, int t0, int kc) {
    ASYNC_COPY16(&qt[buf][lrow][lc], &Qbase[(size_t)(u0 + lrow) * CC + kc + lc]);
    ASYNC_COPY16(&kt[buf][lrow][lc], &Kbase[(size_t)(t0 + lrow) * CC + kc + lc]);
  };

  const v8f vzero = {0.f, 0.f, 0.f, 0.f, 0.f, 0.f, 0.f, 0.f};
  const int NSTAGE = CC / 32;           // 32 K-steps

  for (int t0 = 0; t0 < LL; t0 += 64) {
    v8f acc0 = vzero, acc1 = vzero;

    stage_copy(0, t0, 0);               // prologue
    for (int s = 0; s < NSTAGE; ++s) {
      const int  buf      = s & 1;
      const bool has_next = (s + 1 < NSTAGE);
      if (has_next) stage_copy(buf ^ 1, t0, (s + 1) * 32);
#if HAVE_ASYNC
      if (has_next) ASYNC_WAIT(2);      // in-order: stage s's 2 copies landed
      else          ASYNC_WAIT(0);
#endif
      __syncthreads();                  // all threads' stage-s LDS writes visible

      const v16bf a  = ld_frag_a(&qt[buf][sm * 16][0], lane, 40);
      const v16bf b0 = ld_frag_b(&kt[buf][sn * 32 +  0][0], lane, 40);
      const v16bf b1 = ld_frag_b(&kt[buf][sn * 32 + 16][0], lane, 40);
      acc0 = __builtin_amdgcn_wmma_f32_16x16x32_bf16(false, a, false, b0, (short)0, acc0, false, false);
      acc1 = __builtin_amdgcn_wmma_f32_16x16x32_bf16(false, a, false, b1, (short)0, acc1, false, false);
      __syncthreads();                  // done reading buf before s+2 overwrites
    }

    // Scatter tile into tau histogram: tau = (u - t) mod L
    const int u_base = u0 + sm * 16 + ((lane >> 4) << 3);
    const int t_a    = t0 + sn * 32 + (lane & 15);
    const int t_b    = t_a + 16;
#pragma unroll
    for (int r = 0; r < 8; ++r) {
      const int uu   = u_base + r;
      int tau0 = uu - t_a;  tau0 = ((tau0 % LL) + LL) % LL;
      int tau1 = uu - t_b;  tau1 = ((tau1 % LL) + LL) % LL;
      atomicAdd(&hist[tau0], acc0[r]);
      atomicAdd(&hist[tau1], acc1[r]);
    }
  }
  __syncthreads();
  for (int i = tid; i < LL; i += 256) atomicAdd(&corr[b * LL + i], hist[i]);
}

// ---------------------------------------------------------------------------
__global__ __launch_bounds__(256) void zero_f32(float* __restrict__ p, int n) {
  const int i = blockIdx.x * 256 + threadIdx.x;
  if (i < n) p[i] = 0.f;
}

// ---------------------------------------------------------------------------
// Per-batch top-k(8) over mean corr (scaled 1/C), then softmax of the weights.
// ---------------------------------------------------------------------------
__global__ __launch_bounds__(256) void topk_softmax(
    const float* __restrict__ corr, int* __restrict__ delays,
    float* __restrict__ weights)
{
  __shared__ float sval[LL];
  __shared__ float rmax[256];
  __shared__ int   rarg[256];
  __shared__ float wsel[TOPK];
  __shared__ int   dsel[TOPK];

  const int b = blockIdx.x, tid = threadIdx.x;
  for (int i = tid; i < LL; i += 256) sval[i] = corr[b * LL + i] * (1.0f / (float)CC);
  __syncthreads();

  for (int k = 0; k < TOPK; ++k) {
    float best = -1e30f; int barg = 0;
    for (int i = tid; i < LL; i += 256)
      if (sval[i] > best) { best = sval[i]; barg = i; }
    rmax[tid] = best; rarg[tid] = barg;
    __syncthreads();
    for (int s = 128; s > 0; s >>= 1) {
      if (tid < s && rmax[tid + s] > rmax[tid]) {
        rmax[tid] = rmax[tid + s]; rarg[tid] = rarg[tid + s];
      }
      __syncthreads();
    }
    if (tid == 0) { wsel[k] = rmax[0]; dsel[k] = rarg[0]; sval[rarg[0]] = -1e30f; }
    __syncthreads();
  }

  if (tid == 0) {
    float mx = wsel[0];
    for (int i = 1; i < TOPK; ++i) mx = fmaxf(mx, wsel[i]);
    float e[TOPK], se = 0.f;
    for (int i = 0; i < TOPK; ++i) { e[i] = __expf(wsel[i] - mx); se += e[i]; }
    for (int i = 0; i < TOPK; ++i) {
      weights[b * TOPK + i] = e[i] / se;
      delays[b * TOPK + i]  = dsel[i];
    }
  }
}

// ---------------------------------------------------------------------------
// Time-delay aggregation: agg[b,t,c] = sum_i w[b,i] * V[b,(t+d_i)%L,c]
// One block per (b,t) row; float4 per thread.
// ---------------------------------------------------------------------------
__global__ __launch_bounds__(256) void agg_kernel(
    const float* __restrict__ Vf, const int* __restrict__ delays,
    const float* __restrict__ weights, float* __restrict__ aggf)
{
  const size_t idx4 = (size_t)blockIdx.x * 256 + threadIdx.x;
  const int c  = (int)(idx4 & 255) * 4;          // CC/4 == 256 float4 per row
  const size_t row = idx4 >> 8;
  const int t = (int)(row % LL);
  const int b = (int)(row / LL);

  float ax = 0.f, ay = 0.f, az = 0.f, aw = 0.f;
#pragma unroll
  for (int i = 0; i < TOPK; ++i) {
    const int   d = delays[b * TOPK + i];
    const float w = weights[b * TOPK + i];
    int ts = t + d; if (ts >= LL) ts -= LL;
    const float4 s = *(const float4*)&Vf[((size_t)b * LL + ts) * CC + c];
    ax += w * s.x; ay += w * s.y; az += w * s.z; aw += w * s.w;
  }
  float4 o; o.x = ax; o.y = ay; o.z = az; o.w = aw;
  *(float4*)&aggf[((size_t)b * LL + t) * CC + c] = o;
}

// ---------------------------------------------------------------------------
extern "C" void kernel_launch(void* const* d_in, const int* in_sizes, int n_in,
                              void* d_out, int out_size, void* d_ws, size_t ws_size,
                              hipStream_t stream) {
  const float* queries = (const float*)d_in[0];
  const float* keys    = (const float*)d_in[1];
  const float* values  = (const float*)d_in[2];
  const float* Wq = (const float*)d_in[3];
  const float* bq = (const float*)d_in[4];
  const float* Wk = (const float*)d_in[5];
  const float* bk = (const float*)d_in[6];
  const float* Wv = (const float*)d_in[7];
  const float* bv = (const float*)d_in[8];
  const float* Wo = (const float*)d_in[9];
  const float* bo = (const float*)d_in[10];

  // Workspace layout
  char* ws = (char*)d_ws;
  const size_t szQK  = (size_t)MROWS * CC * sizeof(__bf16);  // 50,331,648
  const size_t szF32 = (size_t)MROWS * CC * sizeof(float);   // 100,663,296
  __bf16* Qb    = (__bf16*)(ws);
  __bf16* Kb    = (__bf16*)(ws + szQK);
  float*  Vf    = (float*) (ws + 2 * szQK);
  float*  aggf  = (float*) (ws + 2 * szQK + szF32);
  float*  corr  = (float*) (ws + 2 * szQK + 2 * szF32);
  int*    dlay  = (int*)   (ws + 2 * szQK + 2 * szF32 + (size_t)BB * LL * 4 + 256);
  float*  wts   = (float*) (ws + 2 * szQK + 2 * szF32 + (size_t)BB * LL * 4 + 512);

  const dim3 blk(256);
  const dim3 gemm_grid(CC / 128, MROWS / 128);   // (8, 192)

  // 1-3) Q/K/V projections (Q,K written as bf16 for the WMMA correlation; V fp32)
  gemm_bias<true ><<<gemm_grid, blk, 0, stream>>>(queries, Wq, bq, Qb, MROWS, CC, CC);
  gemm_bias<true ><<<gemm_grid, blk, 0, stream>>>(keys,    Wk, bk, Kb, MROWS, CC, CC);
  gemm_bias<false><<<gemm_grid, blk, 0, stream>>>(values,  Wv, bv, Vf, MROWS, CC, CC);

  // 4) channel-averaged circular correlation (WMMA Gram + tau histogram)
  zero_f32<<<dim3((BB * LL + 255) / 256), blk, 0, stream>>>(corr, BB * LL);
  corr_kernel<<<dim3(LL / 64, BB), blk, 0, stream>>>(Qb, Kb, corr);

  // 5) top-k delays + softmax weights per batch
  topk_softmax<<<dim3(BB), blk, 0, stream>>>(corr, dlay, wts);

  // 6) time-delay aggregation
  agg_kernel<<<dim3((size_t)MROWS * CC / 4 / 256), blk, 0, stream>>>(Vf, dlay, wts, aggf);

  // 7) output projection -> d_out (fp32)
  gemm_bias<false><<<gemm_grid, blk, 0, stream>>>(aggf, Wo, bo, (float*)d_out, MROWS, CC, CC);
}
